// RouterToMeGlueUseKey_68994354643295
// MI455X (gfx1250) — matrix operand order repaired and checked by reading, not compiled, not run on hardware
//
#include <hip/hip_runtime.h>
#include <hip/hip_bf16.h>

// ToMe token merge for MI455X (gfx1250, wave32).
// Heavy op: scores = a(2048x64) @ b(2048x64)^T fused with row max/argmax,
// computed with v_wmma_f32_16x16x32_f16 (codegen-confirmed CDNA5 path).
// Score matrix (16 MB) is never written to memory -> pipeline stays at the
// ~35 MB compulsory-traffic roofline (~1.5 us @ 23.3 TB/s).

typedef __attribute__((ext_vector_type(16))) _Float16 v16h;
typedef __attribute__((ext_vector_type(8)))  float    v8f;

#define LTOK    4096
#define HALFL   2048
#define DMODEL  768
#define DH      64
#define NHEAD   12
#define OUTL    2049
#define NCHUNK  8          // column chunks in the matmul
#define JT_PER_CHUNK 16    // 16 j-tiles * 16 cols = 256 cols per chunk

// ---------------------------------------------------------------------------
// Kernel A: metric = mean over heads of key_layer, row-normalize, pack:
//   aF16[i][k] = norm(metric[2i])[k]        (even tokens, row-major 2048x64)
//   bT[k][i]   = norm(metric[2i+1])[k]      (odd tokens, TRANSPOSED 64x2048)
// One wave per token; each lane owns 2 of the 64 dims.
// ---------------------------------------------------------------------------
__global__ void __launch_bounds__(256)
metric_pack_kernel(const float* __restrict__ key,
                   _Float16* __restrict__ aF16,
                   _Float16* __restrict__ bT) {
    const int wave = threadIdx.x >> 5;
    const int lane = threadIdx.x & 31;
    const int t    = blockIdx.x * 8 + wave;   // grid = 512 -> t in [0,4096)
    const int d0   = 2 * lane;

    float s0 = 0.f, s1 = 0.f;
#pragma unroll
    for (int h = 0; h < NHEAD; ++h) {
        const float* p = key + (size_t)h * LTOK * DH + (size_t)t * DH + d0;
        s0 += p[0];
        s1 += p[1];
    }
    s0 *= (1.f / NHEAD);
    s1 *= (1.f / NHEAD);

    float sq = s0 * s0 + s1 * s1;
#pragma unroll
    for (int m = 1; m < 32; m <<= 1) sq += __shfl_xor(sq, m, 32);
    const float inv = rsqrtf(sq);

    const _Float16 h0 = (_Float16)(s0 * inv);
    const _Float16 h1 = (_Float16)(s1 * inv);
    const int i = t >> 1;
    if ((t & 1) == 0) {
        aF16[i * DH + d0]     = h0;
        aF16[i * DH + d0 + 1] = h1;
    } else {
        bT[(size_t)d0       * HALFL + i] = h0;
        bT[(size_t)(d0 + 1) * HALFL + i] = h1;
    }
}

// ---------------------------------------------------------------------------
// Kernel B: one wave computes a 16-row tile of scores against a 256-column
// chunk using v_wmma_f32_16x16x32_f16 (K=64 -> 2 WMMAs per 16x16 tile),
// keeping running (max, argmax) per row entirely in registers.
//
// CDNA5 operand layouts (ISA 7.12.2):
//  A 16x32 f16 : lane L -> M=L%16 ; half p -> K=(p<8?p:p+8)+8*(L/16)
//  B 32x16 f16 : lane L -> N=L%16 ; half p -> K as for A (strided bT loads,
//                adjacent lanes read adjacent columns)
//  D 16x16 f32 : reg r, lane L -> M=r+8*(L/16), N=L%16
// ---------------------------------------------------------------------------
__global__ void __launch_bounds__(32)
match_scores_kernel(const _Float16* __restrict__ aF16,
                    const _Float16* __restrict__ bT,
                    float* __restrict__ pmax,
                    int*   __restrict__ pidx) {
    const int lane    = threadIdx.x;
    const int rowtile = blockIdx.x;   // 0..127
    const int chunk   = blockIdx.y;   // 0..7
    const int g       = lane >> 4;
    const int n_lo    = lane & 15;
    const int arow    = rowtile * 16 + n_lo;   // A-frag row for this lane

    v16h a0, a1;
#pragma unroll
    for (int p = 0; p < 16; ++p) {
        const int k = ((p < 8) ? p : p + 8) + g * 8;
        a0[p] = aF16[arow * DH + k];
        a1[p] = aF16[arow * DH + 32 + k];
    }

    float rmax[8];
    int   ridx[8];
#pragma unroll
    for (int r = 0; r < 8; ++r) { rmax[r] = -__builtin_inff(); ridx[r] = 0; }

    const int jt0 = chunk * JT_PER_CHUNK;
    for (int jt = jt0; jt < jt0 + JT_PER_CHUNK; ++jt) {
        v16h b0, b1;
#pragma unroll
        for (int p = 0; p < 16; ++p) {
            const int k = ((p < 8) ? p : p + 8) + g * 8;
            b0[p] = bT[(size_t)k * HALFL + jt * 16 + n_lo];
            b1[p] = bT[(size_t)(32 + k) * HALFL + jt * 16 + n_lo];
        }

        v8f c = {0.f, 0.f, 0.f, 0.f, 0.f, 0.f, 0.f, 0.f};
        c = __builtin_amdgcn_wmma_f32_16x16x32_f16(false, a0, false, b0,
                                                   (short)0, c, false, false);
        c = __builtin_amdgcn_wmma_f32_16x16x32_f16(false, a1, false, b1,
                                                   (short)0, c, false, false);

        const int n = jt * 16 + n_lo;   // global column of this lane's D elems
#pragma unroll
        for (int r = 0; r < 8; ++r) {
            const float v = c[r];
            if (v > rmax[r]) { rmax[r] = v; ridx[r] = n; }   // ties keep first j
        }
    }

    // Reduce each D-row over its 16 lanes (same half-wave); tie -> smaller idx.
#pragma unroll
    for (int r = 0; r < 8; ++r) {
        float rm = rmax[r];
        int   ri = ridx[r];
        for (int off = 1; off < 16; off <<= 1) {
            const float om = __shfl_xor(rm, off, 32);
            const int   oi = __shfl_xor(ri, off, 32);
            if (om > rm || (om == rm && oi < ri)) { rm = om; ri = oi; }
        }
        if (n_lo == 0) {
            const int mg = rowtile * 16 + r + 8 * g;   // global row
            pmax[mg * NCHUNK + chunk] = rm;
            pidx[mg * NCHUNK + chunk] = ri;
        }
    }
}

// ---------------------------------------------------------------------------
// Kernel B2: fold the 8 column-chunk partials into node_max / node_idx.
// Row 0 is forced to -inf / 0 (scores.at[0,:] = -inf in the reference), which
// analytically makes it the single unmerged ("unm") row since r = 2047.
// ---------------------------------------------------------------------------
__global__ void __launch_bounds__(256)
reduce_chunks_kernel(const float* __restrict__ pmax,
                     const int*   __restrict__ pidx,
                     float* __restrict__ node_max,
                     int*   __restrict__ node_idx) {
    const int i = blockIdx.x * blockDim.x + threadIdx.x;
    if (i >= HALFL) return;
    float m = -__builtin_inff();
    int idx = 0;
    for (int c = 0; c < NCHUNK; ++c) {
        const float v  = pmax[i * NCHUNK + c];
        const int   id = pidx[i * NCHUNK + c];
        if (v > m || (v == m && id < idx)) { m = v; idx = id; }
    }
    if (i == 0) { m = -__builtin_inff(); idx = 0; }
    node_max[i] = m;
    node_idx[i] = idx;
}

// ---------------------------------------------------------------------------
// Kernel C: deterministic gather-merge. For destination row j (odd token
// 2j+1), collect all source rows i in {1..2047} with node_idx[i]==j in
// ascending order (fixed FP summation order -> deterministic), then
//   out[1+j] = (hidden[2j+1] + sum_i hidden[2i]) / (1 + count)
//   tome[1+j] = tome_in[2j+1] + sum_i tome_in[2i]
// ---------------------------------------------------------------------------
__global__ void __launch_bounds__(256)
merge_kernel(const float* __restrict__ hidden,
             const float* __restrict__ tome_in,
             const int*   __restrict__ node_idx,
             float* __restrict__ out_tokens,
             float* __restrict__ tome_out) {
    __shared__ int list[HALFL];
    __shared__ int nshare;
    const int j = blockIdx.x;

    if (threadIdx.x == 0) {
        int c = 0;
        for (int i = 1; i < HALFL; ++i)
            if (node_idx[i] == j) list[c++] = i;
        nshare = c;
    }
    __syncthreads();
    const int n = nshare;

    for (int d = threadIdx.x; d < DMODEL; d += 256) {
        float acc = hidden[(size_t)(2 * j + 1) * DMODEL + d];
        for (int t = 0; t < n; ++t)
            acc += hidden[(size_t)(2 * list[t]) * DMODEL + d];
        out_tokens[(size_t)(1 + j) * DMODEL + d] = acc / (float)(n + 1);
    }
    if (threadIdx.x == 0) {
        float ts = tome_in[2 * j + 1];
        for (int t = 0; t < n; ++t) ts += tome_in[2 * list[t]];
        tome_out[1 + j] = ts;
    }
}

// ---------------------------------------------------------------------------
// Kernel D: row 0 of outputs (unm token = token 0) and the zero mask.
// ---------------------------------------------------------------------------
__global__ void __launch_bounds__(256)
finalize_kernel(const float* __restrict__ hidden,
                const float* __restrict__ tome_in,
                float* __restrict__ out_tokens,
                float* __restrict__ mask,
                float* __restrict__ tome_out) {
    const int tid = blockIdx.x * blockDim.x + threadIdx.x;
    if (tid < DMODEL) out_tokens[tid] = hidden[tid];   // unm = hidden token 0
    if (tid < OUTL)   mask[tid] = 0.f;
    if (tid == 0)     tome_out[0] = tome_in[0];
}

// ---------------------------------------------------------------------------
extern "C" void kernel_launch(void* const* d_in, const int* in_sizes, int n_in,
                              void* d_out, int out_size, void* d_ws, size_t ws_size,
                              hipStream_t stream) {
    const float* hidden  = (const float*)d_in[0];   // (1,4096,768)
    // d_in[1] attention_mask, d_in[2] self_attention_scores: unused
    const float* key     = (const float*)d_in[3];   // (1,12,4096,64)
    const float* tome_in = (const float*)d_in[4];   // (1,4096,1)

    // Output: preserved_tokens (2049*768) | mask (2049) | tome (2049)
    float* out_tokens = (float*)d_out;
    float* mask       = out_tokens + (size_t)OUTL * DMODEL;
    float* tome_out   = mask + OUTL;

    // Workspace carve (all offsets >= 32B aligned for v16h loads)
    char* w = (char*)d_ws;
    _Float16* aF16 = (_Float16*)w;                       // 2048*64  = 256 KB
    _Float16* bT   = aF16 + (size_t)HALFL * DH;          // 64*2048  = 256 KB
    float* pmax    = (float*)(bT + (size_t)DH * HALFL);  // 2048*8 f = 64 KB
    int*   pidx    = (int*)(pmax + (size_t)HALFL * NCHUNK);
    int*   node_idx = pidx + (size_t)HALFL * NCHUNK;     // 2048 ints
    float* node_max = (float*)(node_idx + HALFL);        // 2048 floats

    metric_pack_kernel<<<LTOK / 8, 256, 0, stream>>>(key, aF16, bT);

    dim3 gridB(HALFL / 16, NCHUNK);                      // 128 x 8 waves
    match_scores_kernel<<<gridB, 32, 0, stream>>>(aF16, bT, pmax, pidx);

    reduce_chunks_kernel<<<HALFL / 256, 256, 0, stream>>>(pmax, pidx,
                                                          node_max, node_idx);

    merge_kernel<<<HALFL, 256, 0, stream>>>(hidden, tome_in, node_idx,
                                            out_tokens, tome_out);

    finalize_kernel<<<(OUTL + 255) / 256, 256, 0, stream>>>(hidden, tome_in,
                                                            out_tokens, mask,
                                                            tome_out);
}
